// IstftSynthesis_33019708571905
// MI455X (gfx1250) — matrix operand order repaired and checked by reading, not compiled, not run on hardware
//
#include <hip/hip_runtime.h>
#include <hip/hip_bf16.h>
#include <math.h>

// ---------------------------------------------------------------------------
// ISTFT synthesis for MI455X (gfx1250): two f16-WMMA GEMMs fed by the Tensor
// Data Mover (tensor_load_to_lds -> LDS double-buffer -> ds_load fragments),
// plus elementwise glue. Wave32, v_wmma_f32_16x16x32_f16.
// ---------------------------------------------------------------------------

typedef _Float16 half16 __attribute__((ext_vector_type(16)));
typedef _Float16 half8  __attribute__((ext_vector_type(8)));
typedef float    floatx8 __attribute__((ext_vector_type(8)));
typedef unsigned int u32x4 __attribute__((ext_vector_type(4)));
typedef int      i32x4 __attribute__((ext_vector_type(4)));
typedef int      i32x8 __attribute__((ext_vector_type(8)));

#define B_DIM   8
#define C_DIM   512
#define T_DIM   2048
#define NFFT    1024
#define HOP     256
#define O_DIM   1026      // n_fft + 2
#define KHALF   513
#define MPAD1   1088      // gemm1 M padded to 17*64
#define K2PAD   1056      // gemm2 K padded to 33*32
#define LOUT    524288    // (L - n_fft) + 2*pad, L = 1024 + 256*2047
#define PADOUT  128       // hop/2

// ---- workspace layout (bytes, all 64KB-aligned offsets) --------------------
#define OFF_WH   ((size_t)0)                      // f16 [1088][512]      1.06 MB
#define OFF_XT   ((size_t)1114112)                // f16 [8][2048][512]  16.8 MB
#define OFF_Y    ((size_t)17891328)               // f32 [8][1088][2048] 71.3 MB
#define OFF_F    OFF_Y                            // f32 [8][2048][1024] aliases Y
#define OFF_V    ((size_t)89194496)               // f16 [8][2048][1056] 34.6 MB
#define OFF_A2   ((size_t)123797504)              // f16 [1024][1056]     2.2 MB
// total ~126 MB (fits in the 192MB L2 -> GEMM operands stay cache-resident)

#if defined(__has_builtin)
#if __has_builtin(__builtin_amdgcn_tensor_load_to_lds)
#define USE_TDM 1
#endif
#endif
#ifndef USE_TDM
#define USE_TDM 0
#endif

// ---------------------------------------------------------------------------
// TDM: post one 2D-tile descriptor (rows x 64B) global -> LDS. data_size = 8B
// units; tensor dims set huge (tiles are in-bounds by construction, padding in
// ws guarantees it). D# group0: count=1, lds_addr, global_addr, type=2.
#if USE_TDM
__device__ __forceinline__ void tdm_load_tile(const void* gptr, unsigned lds_off,
                                              unsigned rows, unsigned stride_bytes) {
  unsigned long long ga = (unsigned long long)(size_t)gptr;
  u32x4 g0;
  g0.x = 1u;                                        // count = 1 descriptor
  g0.y = lds_off;                                   // LDS byte address
  g0.z = (unsigned)(ga & 0xFFFFFFFFu);              // global_addr[31:0]
  g0.w = (unsigned)((ga >> 32) & 0x01FFFFFFu) | (2u << 30);  // [56:32] | type=2
  const unsigned td0 = 0x7FFFFFFFu, td1 = 0x7FFFFFFFu;   // huge tensor dims
  const unsigned tile_d0 = 64u >> 3;                     // 64B row = 8 units
  unsigned s0 = stride_bytes >> 3;                       // row stride in 8B units
  i32x8 g1;
  g1[0] = (int)(3u << 16);                               // data_size = 3 (8B)
  g1[1] = (int)((td0 & 0xFFFFu) << 16);                  // tensor_dim0 lo
  g1[2] = (int)(((td0 >> 16) & 0xFFFFu) | ((td1 & 0xFFFFu) << 16));
  g1[3] = (int)(((td1 >> 16) & 0xFFFFu) | (tile_d0 << 16));
  g1[4] = (int)(rows & 0xFFFFu);                         // tile_dim1, tile_dim2=0
  g1[5] = (int)s0;                                       // tensor_dim0_stride lo
  g1[6] = 0;
  g1[7] = 0;
  i32x4 z4 = {0, 0, 0, 0};
#if __clang_major__ >= 23
  i32x8 z8 = {0, 0, 0, 0, 0, 0, 0, 0};
  __builtin_amdgcn_tensor_load_to_lds(g0, g1, z4, z4, z8, 0);
#else
  __builtin_amdgcn_tensor_load_to_lds(g0, g1, z4, z4, 0);
#endif
}
__device__ __forceinline__ unsigned lds_off_of(const void* p) {
  return (unsigned)(size_t)p;       // low 32 bits of flat shared addr = LDS byte addr
}
#else
// Fallback: cooperative vector copy (rows x 32 f16) global -> LDS, 128 threads.
__device__ __forceinline__ void coop_copy_tile(_Float16* dst, const _Float16* src,
                                               int rows, int stride_f16) {
  int total = rows * 8;                                  // 8B chunks
  for (int idx = threadIdx.x; idx < total; idx += 128) {
    int r = idx >> 3, c = idx & 7;
    *(unsigned long long*)(dst + r * 32 + c * 4) =
        *(const unsigned long long*)(src + (size_t)r * stride_f16 + c * 4);
  }
}
#endif

// ---------------------------------------------------------------------------
// WMMA fragment loaders from LDS tiles stored [row][32 f16].
// A (16x32): lane l -> M=l&15, h=l>>4; two contiguous 16B chunks at K=8h, 8h+16.
__device__ __forceinline__ half16 frag_a_lds(const _Float16* tile, int row0, int ll, int h) {
  const _Float16* p = tile + (row0 + ll) * 32 + 8 * h;
  union { half16 v; half8 q[2]; } u;
  u.q[0] = *(const half8*)(p);
  u.q[1] = *(const half8*)(p + 16);
  return u.v;
}
// B (32x16) from N-major tile: lane l -> N=l&15; two contiguous 16B chunks at K=8h, 8h+16.
__device__ __forceinline__ half16 frag_b_lds(const _Float16* tile, int col0, int ll, int h) {
  const _Float16* p = tile + (col0 + ll) * 32 + 8 * h;
  union { half16 v; half8 q[2]; } u;
  u.q[0] = *(const half8*)(p);
  u.q[1] = *(const half8*)(p + 16);
  return u.v;
}

// ---------------------------------------------------------------------------
// TDM-fed GEMM core. Block = 4 waves, block tile 64(M) x 256(N); each wave owns
// one 64x64 N-quarter (4x4 WMMA accumulators). Double-buffered LDS, wave 0
// posts next K-step's A/B tiles and retires the current pair with
// s_wait_tensorcnt before the workgroup barrier.
template <int LDA, int NK>
__device__ __forceinline__ void gemm_core(const _Float16* __restrict__ Ablk,
                                          const _Float16* __restrict__ Bblk,
                                          int wave, int ll, int h,
                                          floatx8 (&acc)[4][4]) {
  __shared__ _Float16 As[2][64 * 32];
  __shared__ _Float16 Bs[2][256 * 32];
  bool issuer = (wave == 0);
#if USE_TDM
  if (issuer) {
    tdm_load_tile(Ablk, lds_off_of(&As[0][0]), 64, LDA * 2);
    tdm_load_tile(Bblk, lds_off_of(&Bs[0][0]), 256, LDA * 2);
  }
#else
  coop_copy_tile(&As[0][0], Ablk, 64, LDA);
  coop_copy_tile(&Bs[0][0], Bblk, 256, LDA);
#endif
  for (int ks = 0; ks < NK; ++ks) {
    int cur = ks & 1, nxt = cur ^ 1;
    if (ks + 1 < NK) {
      int k0n = (ks + 1) * 32;
#if USE_TDM
      if (issuer) {
        tdm_load_tile(Ablk + k0n, lds_off_of(&As[nxt][0]), 64, LDA * 2);
        tdm_load_tile(Bblk + k0n, lds_off_of(&Bs[nxt][0]), 256, LDA * 2);
        __builtin_amdgcn_s_wait_tensorcnt((unsigned short)2);  // current pair done
      }
#else
      coop_copy_tile(&As[nxt][0], Ablk + k0n, 64, LDA);
      coop_copy_tile(&Bs[nxt][0], Bblk + k0n, 256, LDA);
#endif
    } else {
#if USE_TDM
      if (issuer) __builtin_amdgcn_s_wait_tensorcnt((unsigned short)0);
#endif
    }
    __syncthreads();                    // publish current buffers to all waves
    half16 af[4], bf[4];
    #pragma unroll
    for (int mi = 0; mi < 4; ++mi) af[mi] = frag_a_lds(As[cur], mi * 16, ll, h);
    #pragma unroll
    for (int ni = 0; ni < 4; ++ni)
      bf[ni] = frag_b_lds(Bs[cur], wave * 64 + ni * 16, ll, h);
    #pragma unroll
    for (int mi = 0; mi < 4; ++mi)
      #pragma unroll
      for (int ni = 0; ni < 4; ++ni)
        acc[mi][ni] = __builtin_amdgcn_wmma_f32_16x16x32_f16(
            false, af[mi], false, bf[ni], (short)0, acc[mi][ni], false, false);
    __syncthreads();                    // buffer safe to overwrite next iter
  }
}

// ---------------------------------------------------------------------------
// 1) W (f32 [1026][512]) -> Wh (f16 [1088][512], zero-padded rows)
__global__ void k_convert_w(const float* __restrict__ W, _Float16* __restrict__ Wh) {
  int i = blockIdx.x * 256 + threadIdx.x;
  if (i >= MPAD1 * C_DIM) return;
  int o = i / C_DIM;
  Wh[i] = (o < O_DIM) ? (_Float16)W[i] : (_Float16)0.0f;
}

// ---------------------------------------------------------------------------
// 2) x (f32 [B][512][2048]) -> xT (f16 [B][2048][512])  (LDS-tiled transpose)
__global__ void k_transpose_x(const float* __restrict__ x, _Float16* __restrict__ xT) {
  __shared__ float tile[32][33];
  int b  = blockIdx.z;
  int t0 = blockIdx.x * 32;
  int c0 = blockIdx.y * 32;
  int tx = threadIdx.x, ty = threadIdx.y;          // blockDim = (32, 8)
  const float* xb = x + (size_t)b * C_DIM * T_DIM;
  #pragma unroll
  for (int i = 0; i < 32; i += 8)
    tile[ty + i][tx] = xb[(size_t)(c0 + ty + i) * T_DIM + t0 + tx];
  __syncthreads();
  _Float16* xTb = xT + (size_t)b * T_DIM * C_DIM;
  #pragma unroll
  for (int i = 0; i < 32; i += 8)
    xTb[(size_t)(t0 + ty + i) * C_DIM + c0 + tx] = (_Float16)tile[tx][ty + i];
}

// ---------------------------------------------------------------------------
// 3) GEMM1: Y[b][o][t] = sum_c Wh[o][c] * xT[b][t][c]   (M=1088,N=2048,K=512)
__global__ void __launch_bounds__(128) k_gemm1(const _Float16* __restrict__ A,
                                               const _Float16* __restrict__ Bx,
                                               float* __restrict__ Y) {
  const int NB = T_DIM / 256, MT = MPAD1 / 64;           // 8, 17
  int bid = blockIdx.x;
  int nb = bid % NB;
  int tmp = bid / NB;
  int mt = tmp % MT, b = tmp / MT;
  int wave = threadIdx.x >> 5;
  int lane = threadIdx.x & 31, ll = lane & 15, h = lane >> 4;
  const _Float16* Ablk = A + (size_t)(mt * 64) * C_DIM;
  const _Float16* Bblk = Bx + ((size_t)b * T_DIM + nb * 256) * C_DIM;
  floatx8 acc[4][4] = {};
  gemm_core<C_DIM, C_DIM / 32>(Ablk, Bblk, wave, ll, h, acc);
  // D layout: VGPR r, lanes 0-15 -> M=r, N=lane; lanes 16-31 -> M=r+8, N=lane-16
  float* Yb = Y + (size_t)b * MPAD1 * T_DIM;
  #pragma unroll
  for (int mi = 0; mi < 4; ++mi)
    #pragma unroll
    for (int ni = 0; ni < 4; ++ni) {
      int row0 = mt * 64 + mi * 16 + 8 * h;
      int col  = nb * 256 + wave * 64 + ni * 16 + ll;
      #pragma unroll
      for (int r = 0; r < 8; ++r)
        Yb[(size_t)(row0 + r) * T_DIM + col] = acc[mi][ni][r];
    }
}

// ---------------------------------------------------------------------------
// 4) Activation: mag/phase -> stacked [real ; imag] as f16, N-major, K padded.
__global__ void k_activation(const float* __restrict__ Y, const float* __restrict__ bias,
                             _Float16* __restrict__ V) {
  const int KK = 543;                     // 513 live + 30 pad writers
  int i = blockIdx.x * 256 + threadIdx.x;
  if (i >= B_DIM * T_DIM * KK) return;
  int kk = i % KK, tmp = i / KK;
  int t = tmp % T_DIM, b = tmp / T_DIM;
  _Float16* Vb = V + ((size_t)b * T_DIM + t) * K2PAD;
  if (kk >= KHALF) { Vb[KHALF + kk] = (_Float16)0.0f; return; }
  const float* Yb = Y + (size_t)b * MPAD1 * T_DIM;
  float mag = Yb[(size_t)kk * T_DIM + t] + bias[kk];
  float ph  = Yb[(size_t)(KHALF + kk) * T_DIM + t] + bias[KHALF + kk];
  float amp = __expf(fminf(mag, 4.0f));
  float s, c;
  __sincosf(ph, &s, &c);
  Vb[kk]         = (_Float16)(amp * c);
  Vb[KHALF + kk] = (_Float16)(amp * s);
}

// ---------------------------------------------------------------------------
// 5) Windowed IDFT matrix (exact phase via (k*n) mod 1024):
//    A2[n][k<513]       =  win[n]/N * a_k * cos(2pi k n/N),  a_0=a_512=1 else 2
//    A2[n][513<=k<1026] = -win[n]/N * 2   * sin(2pi k' n/N)
__global__ void k_build_a2(const float* __restrict__ win, _Float16* __restrict__ A2) {
  int i = blockIdx.x * 256 + threadIdx.x;
  if (i >= NFFT * K2PAD) return;
  int k = i % K2PAD, n = i / K2PAD;
  const float step = 6.283185307179586f / (float)NFFT;
  float w = win[n] * (1.0f / (float)NFFT);
  float v = 0.0f;
  if (k < KHALF) {
    float coef = (k == 0 || k == 512) ? 1.0f : 2.0f;
    int ph = (k * n) & (NFFT - 1);
    v = w * coef * __cosf((float)ph * step);
  } else if (k < O_DIM) {
    int kp = k - KHALF;
    int ph = (kp * n) & (NFFT - 1);
    v = -w * 2.0f * __sinf((float)ph * step);
  }
  A2[i] = (_Float16)v;
}

// ---------------------------------------------------------------------------
// 6) GEMM2: frames^T[b][t][n] = sum_k A2[n][k] * V[b][t][k]  (M=1024,K=1056)
//    Transposed store: each accumulator is 8 contiguous f32 per lane -> 2xb128.
__global__ void __launch_bounds__(128) k_gemm2(const _Float16* __restrict__ A,
                                               const _Float16* __restrict__ Bv,
                                               float* __restrict__ F) {
  const int NB = T_DIM / 256, MT = NFFT / 64;            // 8, 16
  int bid = blockIdx.x;
  int nb = bid % NB;
  int tmp = bid / NB;
  int mt = tmp % MT, b = tmp / MT;
  int wave = threadIdx.x >> 5;
  int lane = threadIdx.x & 31, ll = lane & 15, h = lane >> 4;
  const _Float16* Ablk = A + (size_t)(mt * 64) * K2PAD;
  const _Float16* Bblk = Bv + ((size_t)b * T_DIM + nb * 256) * K2PAD;
  floatx8 acc[4][4] = {};
  gemm_core<K2PAD, K2PAD / 32>(Ablk, Bblk, wave, ll, h, acc);
  float* Fb = F + (size_t)b * T_DIM * NFFT;
  #pragma unroll
  for (int mi = 0; mi < 4; ++mi)
    #pragma unroll
    for (int ni = 0; ni < 4; ++ni) {
      int t  = nb * 256 + wave * 64 + ni * 16 + ll;
      int m0 = mt * 64 + mi * 16 + 8 * h;       // rows m0..m0+7 contiguous
      *(floatx8*)(Fb + (size_t)t * NFFT + m0) = acc[mi][ni];
    }
}

// ---------------------------------------------------------------------------
// 7) Overlap-add + Hann^2 envelope normalize + crop + pad.
__global__ void k_ola(const float* __restrict__ F, const float* __restrict__ win,
                      float* __restrict__ out) {
  int i = blockIdx.x * 256 + threadIdx.x;
  if (i >= B_DIM * LOUT) return;
  int p = i & (LOUT - 1), b = i >> 19;
  if (p < PADOUT || p >= LOUT - PADOUT) { out[i] = 0.0f; return; }
  int s = p + (NFFT / 2 - PADOUT);              // p - 128 + 512
  const float* Fb = F + (size_t)b * T_DIM * NFFT;
  float acc = 0.0f, env = 0.0f;
  int thi = s >> 8;
  #pragma unroll
  for (int dt = 0; dt < 4; ++dt) {
    int t = thi - dt;
    int n = s - (t << 8);
    if (t >= 0 && t < T_DIM && n < NFFT) {
      acc += Fb[(size_t)t * NFFT + n];
      float w = win[n];
      env += w * w;
    }
  }
  out[i] = acc / (env > 1e-11f ? env : 1.0f);
}

// ---------------------------------------------------------------------------
extern "C" void kernel_launch(void* const* d_in, const int* in_sizes, int n_in,
                              void* d_out, int out_size, void* d_ws, size_t ws_size,
                              hipStream_t stream) {
  const float* x    = (const float*)d_in[0];   // [8][512][2048]
  const float* W    = (const float*)d_in[1];   // [1026][512]
  const float* bias = (const float*)d_in[2];   // [1026]
  const float* win  = (const float*)d_in[3];   // [1024]
  float* out = (float*)d_out;                  // [8][524288]

  char* ws = (char*)d_ws;
  _Float16* Wh = (_Float16*)(ws + OFF_WH);
  _Float16* xT = (_Float16*)(ws + OFF_XT);
  float*    Y  = (float*)   (ws + OFF_Y);
  _Float16* V  = (_Float16*)(ws + OFF_V);
  _Float16* A2 = (_Float16*)(ws + OFF_A2);
  float*    F  = (float*)   (ws + OFF_F);      // aliases Y (Y dead after act.)

  k_convert_w<<<(MPAD1 * C_DIM + 255) / 256, 256, 0, stream>>>(W, Wh);

  dim3 tb(32, 8), tg(T_DIM / 32, C_DIM / 32, B_DIM);
  k_transpose_x<<<tg, tb, 0, stream>>>(x, xT);

  k_build_a2<<<(NFFT * K2PAD + 255) / 256, 256, 0, stream>>>(win, A2);

  k_gemm1<<<B_DIM * (MPAD1 / 64) * (T_DIM / 256), 128, 0, stream>>>(Wh, xT, Y);

  k_activation<<<(B_DIM * T_DIM * 543 + 255) / 256, 256, 0, stream>>>(Y, bias, V);

  k_gemm2<<<B_DIM * (NFFT / 64) * (T_DIM / 256), 128, 0, stream>>>(A2, V, F);

  k_ola<<<(B_DIM * LOUT) / 256, 256, 0, stream>>>(F, win, out);
}
